// ScaledDotProductAttention_27350351741346
// MI455X (gfx1250) — matrix operand order repaired and checked by reading, not compiled, not run on hardware
//
#include <hip/hip_runtime.h>

// ---------------------------------------------------------------------------
// ALiBi attention (non-causal, padding mask), B=1 H=16 S=2048 D=64, fp32 I/O.
// Flash-attention single pass. Transposed formulation:
//   S^T = K * Q^T   (rows = keys, cols = queries)
//   O^T = V^T * P^T (rows = d,    cols = queries)
// so per-query softmax reductions live in a (lane, lane^16) pair and the
// P^T B-fragment needs only one shfl_xor(16). WMMA f16 inputs, f32 accum.
// Softmax is carried in the base-2 domain (log2e folded into scale & slopes);
// exponentials use __builtin_amdgcn_exp2f -> bare v_exp_f32 (flush-to-zero
// below 2^-126 is exactly the desired behavior for probability weights).
// ---------------------------------------------------------------------------

typedef __attribute__((ext_vector_type(16))) _Float16 v16h;
typedef __attribute__((ext_vector_type(8)))  float    v8f;

namespace {
constexpr int NH    = 16;    // heads
constexpr int SEQ   = 2048;  // sequence length
constexpr int HD    = 64;    // head dim
constexpr int KBLK  = 32;    // keys per block (= WMMA K for the PV gemm)
constexpr int NWAVE = 8;     // waves per workgroup
constexpr int QW    = 16;    // query rows per wave
constexpr int QB    = NWAVE * QW;         // 128 q rows per workgroup
constexpr int KT_S  = 72;    // padded f16 row stride for K tile  [key][d]
constexpr int VT_S  = 40;    // padded f16 row stride for V tile  [d][key]
constexpr float LOG2E  = 1.4426950408889634f;
constexpr float SCALE2 = 0.125f * LOG2E;  // (1/sqrt(64)) * log2(e)
constexpr float NEG_BIG = -3.0e38f;       // ~ finfo(f32).min
}

__global__ __launch_bounds__(256) void alibi_flash_attn(
    const float* __restrict__ Qg, const float* __restrict__ Kg,
    const float* __restrict__ Vg, const unsigned char* __restrict__ maskg,
    float* __restrict__ Og)
{
  __shared__ float    adj[SEQ];            // base-2 bias+mask per key (8 KB)
  __shared__ _Float16 Kt[KBLK][KT_S];      // K tile, [key][d], padded (4.5 KB)
  __shared__ _Float16 Vt[HD][VT_S];        // V tile transposed, [d][key] (5 KB)

  const int tid  = threadIdx.x;
  const int lane = tid & 31;
  const int col  = lane & 15;              // N column (query) this lane owns
  const int hi   = lane >> 4;              // 0: lanes 0-15, 1: lanes 16-31
  const int h    = blockIdx.x % NH;
  const int qblk = blockIdx.x / NH;
  const int q0   = qblk * QB + (tid >> 5) * QW;
  const int q    = q0 + col;

  // ---- per-head bias table in log2 domain:
  //      adj[k] = mask[k] ? log2e*slope*(k-(S-1)) : -BIG ---------------------
  const float slope2 = LOG2E * exp2f(-0.5f * (float)(h + 1)); // H=16 pow2
  for (int k = tid; k < SEQ; k += 256)
    adj[k] = maskg[k] ? slope2 * (float)(k - (SEQ - 1)) : NEG_BIG;

  // ---- Q^T B-fragments (held for the whole pass) --------------------------
  // B layout (16-bit, 32x16): lane col = q, half i -> Kdim d = c*32 + hi*16 + i
  v16h bq[2];
  {
    const float* qrow = Qg + ((size_t)(h * SEQ + q)) * HD + hi * 16;
    #pragma unroll
    for (int c = 0; c < 2; ++c) {
      float4 f0 = ((const float4*)(qrow + c * 32))[0];
      float4 f1 = ((const float4*)(qrow + c * 32))[1];
      float4 f2 = ((const float4*)(qrow + c * 32))[2];
      float4 f3 = ((const float4*)(qrow + c * 32))[3];
      bq[c][0]=(_Float16)f0.x; bq[c][1]=(_Float16)f0.y; bq[c][2]=(_Float16)f0.z; bq[c][3]=(_Float16)f0.w;
      bq[c][4]=(_Float16)f1.x; bq[c][5]=(_Float16)f1.y; bq[c][6]=(_Float16)f1.z; bq[c][7]=(_Float16)f1.w;
      bq[c][8]=(_Float16)f2.x; bq[c][9]=(_Float16)f2.y; bq[c][10]=(_Float16)f2.z; bq[c][11]=(_Float16)f2.w;
      bq[c][12]=(_Float16)f3.x; bq[c][13]=(_Float16)f3.y; bq[c][14]=(_Float16)f3.z; bq[c][15]=(_Float16)f3.w;
    }
  }

  v8f  o[4] = {};                // O^T accumulators: d-tile dt, rows dt*16+hi*8+r
  float m_i = NEG_BIG, l_i = 0.0f;   // running max (log2 domain) and sum

  for (int kb = 0; kb < SEQ; kb += KBLK) {
    __syncthreads();             // previous tile fully consumed
    // ---- cooperative stage of K/V block (32x64 f32 -> f16 LDS) ------------
    {
      const int key = tid >> 3;            // 0..31
      const int d0  = (tid & 7) * 8;       // 0,8,...,56 (wave covers 1KB runs)
      const float* kp = Kg + ((size_t)(h * SEQ + kb + key)) * HD + d0;
      const float* vp = Vg + ((size_t)(h * SEQ + kb + key)) * HD + d0;
      float4 k0 = ((const float4*)kp)[0], k1 = ((const float4*)kp)[1];
      float4 v0 = ((const float4*)vp)[0], v1 = ((const float4*)vp)[1];
      union { _Float16 e[8]; uint4 u; } pk;
      pk.e[0]=(_Float16)k0.x; pk.e[1]=(_Float16)k0.y; pk.e[2]=(_Float16)k0.z; pk.e[3]=(_Float16)k0.w;
      pk.e[4]=(_Float16)k1.x; pk.e[5]=(_Float16)k1.y; pk.e[6]=(_Float16)k1.z; pk.e[7]=(_Float16)k1.w;
      *(uint4*)&Kt[key][d0] = pk.u;                      // 16B aligned
      Vt[d0+0][key]=(_Float16)v0.x; Vt[d0+1][key]=(_Float16)v0.y;
      Vt[d0+2][key]=(_Float16)v0.z; Vt[d0+3][key]=(_Float16)v0.w;
      Vt[d0+4][key]=(_Float16)v1.x; Vt[d0+5][key]=(_Float16)v1.y;
      Vt[d0+6][key]=(_Float16)v1.z; Vt[d0+7][key]=(_Float16)v1.w;
    }
    __syncthreads();

    // ---- GEMM1: S^T tiles (rows = keys kb+t*16+hi*8+r, col = q) -----------
    v8f st[2];
    #pragma unroll
    for (int t = 0; t < 2; ++t) {
      v8f acc = {};
      #pragma unroll
      for (int c = 0; c < 2; ++c) {
        // A layout (16-bit, 16x32): lane row = key, halves: d in
        //   {b..b+15}, b = c*32 + hi*16
        const _Float16* krow = &Kt[t * 16 + col][c * 32 + hi * 16];
        v16h a;
        ((uint4*)&a)[0] = *(const uint4*)(krow);
        ((uint4*)&a)[1] = *(const uint4*)(krow + 8);
        acc = __builtin_amdgcn_wmma_f32_16x16x32_f16(
                  false, a, false, bq[c], (short)0, acc, false, false);
      }
      st[t] = acc;
    }

    // ---- scale + bias/mask (log2 domain), online softmax over block -------
    float vals[2][8];
    #pragma unroll
    for (int t = 0; t < 2; ++t) {
      const float4* a4 = (const float4*)&adj[kb + t * 16 + hi * 8];
      float4 A0 = a4[0], A1 = a4[1];
      float aa[8] = {A0.x,A0.y,A0.z,A0.w,A1.x,A1.y,A1.z,A1.w};
      #pragma unroll
      for (int r = 0; r < 8; ++r) vals[t][r] = st[t][r] * SCALE2 + aa[r];
    }
    float bm = NEG_BIG;
    #pragma unroll
    for (int t = 0; t < 2; ++t)
      #pragma unroll
      for (int r = 0; r < 8; ++r) bm = fmaxf(bm, vals[t][r]);
    bm = fmaxf(bm, __shfl_xor(bm, 16));          // lane pair holds all 32 keys
    const float mnew = fmaxf(m_i, bm);
    // mnew is finite from iteration 0 (key 0 is always valid); masked scores
    // give v_exp_f32(-huge) which flushes to exactly 0.

    float pv[2][8]; float bsum = 0.0f;
    #pragma unroll
    for (int t = 0; t < 2; ++t)
      #pragma unroll
      for (int r = 0; r < 8; ++r) {
        float e = __builtin_amdgcn_exp2f(vals[t][r] - mnew); // bare v_exp_f32
        pv[t][r] = e; bsum += e;
      }
    bsum += __shfl_xor(bsum, 16);
    const float alpha = __builtin_amdgcn_exp2f(m_i - mnew);
    l_i = l_i * alpha + bsum;
    m_i = mnew;
    #pragma unroll
    for (int dt = 0; dt < 4; ++dt)
      #pragma unroll
      for (int r = 0; r < 8; ++r) o[dt][r] *= alpha;

    // ---- build P^T B-fragment: lane col = q, halves = keys ----------------
    // lane<16 needs keys 0..15; lane>=16 needs 16..31; one xor-16 exchange.
    v16h bp;
    #pragma unroll
    for (int r = 0; r < 8; ++r) {
      float send = hi ? pv[0][r] : pv[1][r];
      float recv = __shfl_xor(send, 16);
      float e0 = hi ? recv : pv[0][r];     // half r     (key base+0..7)
      float e1 = hi ? pv[1][r] : recv;     // half 8+r   (key base+8..15)
      bp[r]     = (_Float16)e0;
      bp[8 + r] = (_Float16)e1;
    }

    // ---- GEMM2: O^T += V^T * P^T ------------------------------------------
    #pragma unroll
    for (int dt = 0; dt < 4; ++dt) {
      // A fragment: lane row = d = dt*16+col; halves: keys {b..b+15}, b = hi*16
      const _Float16* vrow = &Vt[dt * 16 + col][hi * 16];
      v16h av;
      ((uint4*)&av)[0] = *(const uint4*)(vrow);
      ((uint4*)&av)[1] = *(const uint4*)(vrow + 8);
      o[dt] = __builtin_amdgcn_wmma_f32_16x16x32_f16(
                  false, av, false, bp, (short)0, o[dt], false, false);
    }
  }

  // ---- epilogue: O[h][q][d] = O^T / l ---------------------------------------
  const float inv = 1.0f / l_i;            // key 0 always valid -> l_i > 0
  float* orow = Og + ((size_t)(h * SEQ + q)) * HD + hi * 8;
  #pragma unroll
  for (int dt = 0; dt < 4; ++dt) {
    float4 r0 = { o[dt][0]*inv, o[dt][1]*inv, o[dt][2]*inv, o[dt][3]*inv };
    float4 r1 = { o[dt][4]*inv, o[dt][5]*inv, o[dt][6]*inv, o[dt][7]*inv };
    ((float4*)(orow + dt * 16))[0] = r0;
    ((float4*)(orow + dt * 16))[1] = r1;
  }
}

extern "C" void kernel_launch(void* const* d_in, const int* in_sizes, int n_in,
                              void* d_out, int out_size, void* d_ws, size_t ws_size,
                              hipStream_t stream) {
  (void)in_sizes; (void)n_in; (void)out_size; (void)d_ws; (void)ws_size;
  const float* Q = (const float*)d_in[0];
  const float* K = (const float*)d_in[1];
  const float* V = (const float*)d_in[2];
  const unsigned char* M = (const unsigned char*)d_in[3];  // jnp bool -> 1 byte
  float* O = (float*)d_out;
  dim3 grid(NH * (SEQ / QB));   // 16 heads x 16 query blocks = 256 WGs
  dim3 block(256);              // 8 wave32s
  alibi_flash_attn<<<grid, block, 0, stream>>>(Q, K, V, M, O);
}
